// PillarFeatureNet_30073361007074
// MI455X (gfx1250) — matrix operand order repaired and checked by reading, not compiled, not run on hardware
//
#include <hip/hip_runtime.h>

// PillarFeatureNet fused kernel for gfx1250 (MI455X), wave32.
// Setup kernel: fold BN into W (W' = W*gamma/sqrt(var+eps), b' = beta - mu*gamma/sqrt(var+eps)),
//   pre-convert to f16 in WMMA B-fragment word layout -> __device__ globals.
// Main kernel: one wave = one pillar per iteration (grid-stride, ~4 pillars/wave):
//   augment 4->9 ch, mask, [32x9]x[9x64] GEMM via 8x v_wmma_f32_16x16x32_f16
//   (K padded 9->32), max over 32 points via VGPR max + 1 cross-half shuffle,
//   bias+ReLU (commute with max), 2 full-wave coalesced stores.
// Bandwidth-bound (~31 MB traffic); no LDS; all cross-lane traffic via shfl_xor.

typedef __attribute__((ext_vector_type(16))) _Float16 v16h;
typedef __attribute__((ext_vector_type(8)))  float    v8f;
typedef __attribute__((ext_vector_type(8)))  unsigned v8u;

#define VOXEL  0.2f
#define XOFF   0.1f      //  VX/2 + 0.0
#define YOFF  -39.9f     //  VY/2 - 40.0
#define BNEPS  1e-3f

// Folded, f16-packed weights: column n, words 0..4 = half2(K0,K1)..half2(K8,0),
// words 5..7 = 0 (K rows 9..15 of the B fragment's lower half).
__device__ unsigned g_bfrag[64 * 8];
__device__ float    g_bias[64];

static __device__ __forceinline__ unsigned pkh2(float lo, float hi) {
#if __has_builtin(__builtin_amdgcn_cvt_pkrtz)
    auto h = __builtin_amdgcn_cvt_pkrtz(lo, hi);  // v_cvt_pk_rtz_f16_f32, __fp16x2
    return __builtin_bit_cast(unsigned, h);
#else
    union { _Float16 h[2]; unsigned u; } t;
    t.h[0] = (_Float16)lo; t.h[1] = (_Float16)hi;
    return t.u;
#endif
}

static __device__ __forceinline__ float wave_sum32(float v) {
    v += __shfl_xor(v, 16, 32);
    v += __shfl_xor(v,  8, 32);
    v += __shfl_xor(v,  4, 32);
    v += __shfl_xor(v,  2, 32);
    v += __shfl_xor(v,  1, 32);
    return v;
}

__global__ __launch_bounds__(64) void pfn_setup_kernel(
        const float* __restrict__ W,      // [9,64]
        const float* __restrict__ gamma,
        const float* __restrict__ beta,
        const float* __restrict__ rmean,
        const float* __restrict__ rvar)
{
    const int n = threadIdx.x;            // 0..63 output channels
    const float is = gamma[n] / sqrtf(rvar[n] + BNEPS);
    g_bias[n] = beta[n] - rmean[n] * is;
    unsigned w[8];
    w[0] = pkh2(W[0 * 64 + n] * is, W[1 * 64 + n] * is);
    w[1] = pkh2(W[2 * 64 + n] * is, W[3 * 64 + n] * is);
    w[2] = pkh2(W[4 * 64 + n] * is, W[5 * 64 + n] * is);
    w[3] = pkh2(W[6 * 64 + n] * is, W[7 * 64 + n] * is);
    w[4] = pkh2(W[8 * 64 + n] * is, 0.0f);
    w[5] = w[6] = w[7] = 0u;
    #pragma unroll
    for (int i = 0; i < 8; ++i) g_bfrag[n * 8 + i] = w[i];
}

__global__ __launch_bounds__(256) void pfn_wmma_kernel(
        const float* __restrict__ feat,   // [N,32,4]
        const int*   __restrict__ nvox,   // [N]
        const int*   __restrict__ coors,  // [N,4] (b,z,y,x)
        float*       __restrict__ out,    // [N,64]
        int N, int stride)                // stride = total waves in grid
{
    const int lane = threadIdx.x & 31;
    const int half = lane >> 4;           // 0: lanes 0-15, 1: lanes 16-31
    const int l16  = lane & 15;

    int p = __builtin_amdgcn_readfirstlane(
                blockIdx.x * (blockDim.x >> 5) + (threadIdx.x >> 5));
    if (p >= N) return;                   // wave-uniform

    // ---- B fragments from precomputed globals (L2-resident, 2.3 KB) ----
    // B 32x16, this code's contiguous k order: VGPR v, lanes 0-15 hold K=2v,2v+1 (col N=lane);
    // lanes 16-31 hold K=16+2v,17+2v -> all zero here (K<=8).
    v16h bfrag[4];
    #pragma unroll
    for (int t = 0; t < 4; ++t) {
        v8u bu = {};
        if (half == 0) {
            const unsigned* src = &g_bfrag[(16 * t + l16) * 8];
            const uint4 lo = *reinterpret_cast<const uint4*>(src);
            bu[0] = lo.x; bu[1] = lo.y; bu[2] = lo.z; bu[3] = lo.w;
            bu[4] = src[4];
        }
        bfrag[t] = __builtin_bit_cast(v16h, bu);
    }
    // this lane stores columns 32*half + l16 and +16
    const float biasS0 = g_bias[32 * half + l16];
    const float biasS1 = g_bias[32 * half + 16 + l16];

    // ---- software-pipelined pillar loop: lane L <-> point L ----
    float4 f   = reinterpret_cast<const float4*>(feat)[p * 32 + lane];
    int    nv  = nvox[p];
    int2   cyx = *reinterpret_cast<const int2*>(coors + p * 4 + 2); // (y, x)

    for (;;) {
        const int  pn = p + stride;
        const bool hn = pn < N;           // wave-uniform
        float4 fn = {}; int nvn = 0; int2 cn = {};
        if (hn) {                         // prefetch next pillar
            fn  = reinterpret_cast<const float4*>(feat)[pn * 32 + lane];
            nvn = nvox[pn];
            cn  = *reinterpret_cast<const int2*>(coors + pn * 4 + 2);
        }

        // mean over ALL 32 points / num_voxels (reference semantics)
        const float inv = 1.0f / (float)nv;
        const float mx = wave_sum32(f.x) * inv;
        const float my = wave_sum32(f.y) * inv;
        const float mz = wave_sum32(f.z) * inv;
        const float cxo = (float)cyx.y * VOXEL + XOFF;
        const float cyo = (float)cyx.x * VOXEL + YOFF;

        // 9 channels packed as half2 words (K0..K8); mask on packed words
        unsigned pk0 = pkh2(f.x, f.y);
        unsigned pk1 = pkh2(f.z, f.w);
        unsigned pk2 = pkh2(f.x - mx, f.y - my);
        unsigned pk3 = pkh2(f.z - mz, f.x - cxo);
        unsigned pk4 = pkh2(f.y - cyo, 0.0f);
        if (lane >= nv) { pk0 = 0u; pk1 = 0u; pk2 = 0u; pk3 = 0u; pk4 = 0u; }

        // cross-half copies (lane L gets lane L^16's words)
        const unsigned xp0 = (unsigned)__shfl_xor((int)pk0, 16, 32);
        const unsigned xp1 = (unsigned)__shfl_xor((int)pk1, 16, 32);
        const unsigned xp2 = (unsigned)__shfl_xor((int)pk2, 16, 32);
        const unsigned xp3 = (unsigned)__shfl_xor((int)pk3, 16, 32);
        const unsigned xp4 = (unsigned)__shfl_xor((int)pk4, 16, 32);

        // 16-bit A 16x32, same contiguous k order as B: lanes 0-15 VGPR0-4 = K0..9 (row M=lane);
        // lanes 16-31 (K16..31) -> zero.
        v8u a0u = {};                     // M-tile 0: points 0..15
        v8u a1u = {};                     // M-tile 1: points 16..31
        if (half == 0) {
            a0u[0] = pk0; a0u[1] = pk1; a0u[2] = pk2; a0u[3] = pk3; a0u[4] = pk4;
            a1u[0] = xp0; a1u[1] = xp1; a1u[2] = xp2; a1u[3] = xp3; a1u[4] = xp4;
        }
        const v16h a0 = __builtin_bit_cast(v16h, a0u);
        const v16h a1 = __builtin_bit_cast(v16h, a1u);

        // 8 WMMAs; per-lane max over row-VGPRs (bias+ReLU commute with max)
        float mv[4];
        #pragma unroll
        for (int t = 0; t < 4; ++t) {
            v8f z = {};
            v8f d0 = __builtin_amdgcn_wmma_f32_16x16x32_f16(
                         false, a0, false, bfrag[t], (short)0, z, false, false);
            v8f d1 = __builtin_amdgcn_wmma_f32_16x16x32_f16(
                         false, a1, false, bfrag[t], (short)0, z, false, false);
            float m = fmaxf(d0[0], d1[0]);
            #pragma unroll
            for (int v = 1; v < 8; ++v) m = fmaxf(m, fmaxf(d0[v], d1[v]));
            mv[t] = m;
        }

        // cross-half merge only for the 2 columns this lane stores:
        // lane (h,l16) stores cols 32h+l16 (tile 2h) and 32h+16+l16 (tile 2h+1);
        // it provides the tiles the opposite half needs.
        const float send0 = (half == 0) ? mv[2] : mv[0];
        const float send1 = (half == 0) ? mv[3] : mv[1];
        const float got0  = __shfl_xor(send0, 16, 32);
        const float got1  = __shfl_xor(send1, 16, 32);
        const float mine0 = (half == 0) ? mv[0] : mv[2];
        const float mine1 = (half == 0) ? mv[1] : mv[3];
        const float o0 = fmaxf(fmaxf(mine0, got0) + biasS0, 0.0f);
        const float o1 = fmaxf(fmaxf(mine1, got1) + biasS1, 0.0f);

        float* op = out + p * 64 + 32 * half + l16;
        op[0]  = o0;                      // full-wave coalesced b32 stores
        op[16] = o1;

        if (!hn) break;
        p = pn; f = fn; nv = nvn; cyx = cn;
    }
}

extern "C" void kernel_launch(void* const* d_in, const int* in_sizes, int n_in,
                              void* d_out, int out_size, void* d_ws, size_t ws_size,
                              hipStream_t stream) {
    const float* feat  = (const float*)d_in[0];
    const int*   nvoxv = (const int*)  d_in[1];
    const int*   coors = (const int*)  d_in[2];
    const float* W     = (const float*)d_in[3];
    const float* gamma = (const float*)d_in[4];
    const float* beta  = (const float*)d_in[5];
    const float* rmean = (const float*)d_in[6];
    const float* rvar  = (const float*)d_in[7];
    float* out = (float*)d_out;

    const int N = in_sizes[1];            // one num_voxels entry per pillar

    pfn_setup_kernel<<<1, 64, 0, stream>>>(W, gamma, beta, rmean, rvar);

    const int wavesPerBlock = 256 / 32;   // 8 waves per workgroup
    int blocks = (N + wavesPerBlock - 1) / wavesPerBlock;
    if (blocks > 1250) blocks = 1250;     // ~10000 waves: ~4 pillars/wave,
                                          // amortizes B-frag load, keeps
                                          // plenty of memory parallelism
    const int stride = blocks * wavesPerBlock;

    pfn_wmma_kernel<<<blocks, 256, 0, stream>>>(
        feat, nvoxv, coors, out, N, stride);
}
